// GraphConvInteraction_78297253806170
// MI455X (gfx1250) — matrix operand order repaired and checked by reading, not compiled, not run on hardware
//
#include <hip/hip_runtime.h>

// ---------------------------------------------------------------------------
// GraphConvInteraction on MI455X (gfx1250, wave32, WMMA).
//
// agg_i = relu((W_self - W_inter) x_i + W_inter xsum + (b_self + 3 b_inter))
// out_i = x_i + W_gx x_i + W_ga agg_i + b_gate,  W_gate = [W_gx | W_ga]
//
// All matmuls run as v_wmma_f32_16x16x32_f16 with 3-term hi/lo f16 split
// (hi*hi + hi*lo + lo*hi) for near-fp32 accuracy.  HBM traffic 512 MiB
// (~23 us @ 23.3 TB/s); ~100 GFLOP of f16 WMMA -- roughly balanced.
// ---------------------------------------------------------------------------

typedef __attribute__((ext_vector_type(16))) _Float16 v16h;
typedef __attribute__((ext_vector_type(8)))  float    v8f;

union AF { v16h h; unsigned int u[8]; };
union PK { _Float16 h[2]; unsigned int u; };

static __device__ __forceinline__ unsigned int pack2(_Float16 a, _Float16 b) {
  PK p; p.h[0] = a; p.h[1] = b; return p.u;
}

#define WMMAF16(a, b, c) \
  __builtin_amdgcn_wmma_f32_16x16x32_f16(false, (a), false, (b), (short)0, (c), false, false)

#define B_    16
#define C_    64
#define HW_   16384
#define CHW_  (C_ * HW_)
#define BCHW_ ((long long)B_ * CHW_)
#define PX    32                       // pixels per workgroup tile
#define NTILE (B_ * HW_ / PX)          // 8192 tiles

// Dynamic LDS layout (bytes)
#define AGG_STRIDE 17
#define OFF_WC1H 0                         // 128x32 packed f16-pairs (hi)
#define OFF_WC1L (OFF_WC1H + 128*32*4)     // 128x32 (lo)
#define OFF_WINH (OFF_WC1L + 128*32*4)     // 64x32 W_inter hi
#define OFF_WINL (OFF_WINH + 64*32*4)      // 64x32 W_inter lo
#define OFF_WGAH (OFF_WINL + 64*32*4)      // 64x32 W_ga hi
#define OFF_WGAL (OFF_WGAH + 64*32*4)      // 64x32 W_ga lo
#define OFF_BA   (OFF_WGAL + 64*32*4)      // 64 f32: b_self + 3 b_inter
#define OFF_BG   (OFF_BA + 64*4)           // 64 f32: b_gate
#define OFF_XT   (OFF_BG + 64*4)           // 4 * 64 * PX f32 pixel tiles
#define OFF_AGG  (OFF_XT + 4*64*PX*4)      // 8 waves * 64*AGG_STRIDE f32
#define SMEM_BYTES (OFF_AGG + 8*64*AGG_STRIDE*4)   // ~130.5 KiB < 320 KiB/WGP

// A-fragment (16x32 f16, contiguous k per half-wave): lane l -> M = l%16,
// VGPR v -> K pair { 2v + 16*(l/16) } (+32*kc), stored packed.
static __device__ __forceinline__ void loadA(const unsigned int* __restrict__ wp,
                                             int rowbase, int kc, int mlane, int h16,
                                             AF& f) {
#pragma unroll
  for (int v = 0; v < 8; ++v) {
    int kp = v + 8 * h16 + 16 * kc;
    f.u[v] = wp[(rowbase + mlane) * 32 + kp];
  }
}

__global__ void __launch_bounds__(256, 1)
gci_kernel(const float* __restrict__ x0, const float* __restrict__ x1,
           const float* __restrict__ x2, const float* __restrict__ x3,
           const float* __restrict__ w_inter, const float* __restrict__ b_inter,
           const float* __restrict__ w_self,  const float* __restrict__ b_self,
           const float* __restrict__ w_gate,  const float* __restrict__ b_gate,
           float* __restrict__ out) {
  extern __shared__ __align__(16) char smem[];
  unsigned int* wc1h = (unsigned int*)(smem + OFF_WC1H);
  unsigned int* wc1l = (unsigned int*)(smem + OFF_WC1L);
  unsigned int* winh = (unsigned int*)(smem + OFF_WINH);
  unsigned int* winl = (unsigned int*)(smem + OFF_WINL);
  unsigned int* wgah = (unsigned int*)(smem + OFF_WGAH);
  unsigned int* wgal = (unsigned int*)(smem + OFF_WGAL);
  float* ba  = (float*)(smem + OFF_BA);
  float* bg  = (float*)(smem + OFF_BG);
  float* xt  = (float*)(smem + OFF_XT);    // [img][c][p], p-stride PX
  float* agg = (float*)(smem + OFF_AGG);

  const int tid = threadIdx.x;

  // ---- one-time per block: convert weights to packed f16 hi/lo in LDS ----
  for (int idx = tid; idx < 128 * 32; idx += 256) {
    int r = idx >> 5, k = (idx & 31) * 2;
    float a0, a1;
    if (r < 64) {
      a0 = w_self[r * 64 + k]     - w_inter[r * 64 + k];
      a1 = w_self[r * 64 + k + 1] - w_inter[r * 64 + k + 1];
    } else {
      a0 = w_gate[(r - 64) * 128 + k];
      a1 = w_gate[(r - 64) * 128 + k + 1];
    }
    _Float16 h0 = (_Float16)a0, h1 = (_Float16)a1;
    wc1h[idx] = pack2(h0, h1);
    wc1l[idx] = pack2((_Float16)(a0 - (float)h0), (_Float16)(a1 - (float)h1));
  }
  for (int idx = tid; idx < 64 * 32; idx += 256) {
    int r = idx >> 5, k = (idx & 31) * 2;
    float a0 = w_inter[r * 64 + k], a1 = w_inter[r * 64 + k + 1];
    _Float16 h0 = (_Float16)a0, h1 = (_Float16)a1;
    winh[idx] = pack2(h0, h1);
    winl[idx] = pack2((_Float16)(a0 - (float)h0), (_Float16)(a1 - (float)h1));
    float g0 = w_gate[r * 128 + 64 + k], g1 = w_gate[r * 128 + 64 + k + 1];
    _Float16 gh0 = (_Float16)g0, gh1 = (_Float16)g1;
    wgah[idx] = pack2(gh0, gh1);
    wgal[idx] = pack2((_Float16)(g0 - (float)gh0), (_Float16)(g1 - (float)gh1));
  }
  if (tid < 64) {
    ba[tid] = b_self[tid] + 3.0f * b_inter[tid];
    bg[tid] = b_gate[tid];
  }
  __syncthreads();

  const int wave = tid >> 5;
  const int lane = tid & 31;
  const int img  = wave & 3;           // which of x1..x4 this wave owns
  const int n0   = (wave >> 2) * 16;   // pixel subtile within the 32-px tile
  const int n    = lane & 15;          // column / M-lane
  const int h16  = lane >> 4;          // upper half-wave flag
  float* aggw = agg + wave * (64 * AGG_STRIDE);

  for (int tile = blockIdx.x; tile < NTILE; tile += gridDim.x) {
    const int b  = tile / (HW_ / PX);
    const int p0 = (tile % (HW_ / PX)) * PX;
    const long long gbase = (long long)b * CHW_ + p0;

    // ---- cooperative x tile load: 4 images x 64 ch x 32 px (float4) ----
    for (int idx = tid; idx < 64 * (PX / 4); idx += 256) {
      int c = idx >> 3, p4 = (idx & 7) * 4;
      long long g = gbase + (long long)c * HW_ + p4;
      int l = c * PX + p4;
      *(float4*)(xt + 0 * 64 * PX + l) = *(const float4*)(x0 + g);
      *(float4*)(xt + 1 * 64 * PX + l) = *(const float4*)(x1 + g);
      *(float4*)(xt + 2 * 64 * PX + l) = *(const float4*)(x2 + g);
      *(float4*)(xt + 3 * 64 * PX + l) = *(const float4*)(x3 + g);
    }
    __syncthreads();

    const float* xti = xt + img * 64 * PX;

    v8f acc1[8];   // [Wc1 @ x_i]: rows 0..63 pre-agg, rows 64..127 W_gx@x_i
    v8f accs[4];   // W_inter @ xsum
#pragma unroll
    for (int m = 0; m < 8; ++m)
#pragma unroll
      for (int e = 0; e < 8; ++e) acc1[m][e] = 0.0f;
#pragma unroll
    for (int m = 0; m < 4; ++m)
#pragma unroll
      for (int e = 0; e < 8; ++e) accs[m][e] = 0.0f;

    // ---- stage 1: Wc1 @ x_i  and  W_inter @ xsum ----
#pragma unroll
    for (int kc = 0; kc < 2; ++kc) {
      AF bxh, bxl, bsh, bsl;
#pragma unroll
      for (int v = 0; v < 8; ++v) {
        // B fragment (32x16): VGPR v, lanes 0-15 hold K=2v,2v+1; lanes 16-31 K+16
        int k = kc * 32 + h16 * 16 + 2 * v;
        int o0 = k * PX + n0 + n, o1 = (k + 1) * PX + n0 + n;
        float f0 = xti[o0], f1 = xti[o1];
        float s0 = xt[o0] + xt[64 * PX + o0] + xt[2 * 64 * PX + o0] + xt[3 * 64 * PX + o0];
        float s1 = xt[o1] + xt[64 * PX + o1] + xt[2 * 64 * PX + o1] + xt[3 * 64 * PX + o1];
        _Float16 fh0 = (_Float16)f0, fh1 = (_Float16)f1;
        bxh.u[v] = pack2(fh0, fh1);
        bxl.u[v] = pack2((_Float16)(f0 - (float)fh0), (_Float16)(f1 - (float)fh1));
        _Float16 sh0 = (_Float16)s0, sh1 = (_Float16)s1;
        bsh.u[v] = pack2(sh0, sh1);
        bsl.u[v] = pack2((_Float16)(s0 - (float)sh0), (_Float16)(s1 - (float)sh1));
      }
#pragma unroll
      for (int mt = 0; mt < 8; ++mt) {
        AF ah, al;
        loadA(wc1h, mt * 16, kc, n, h16, ah);
        loadA(wc1l, mt * 16, kc, n, h16, al);
        acc1[mt] = WMMAF16(ah.h, bxh.h, acc1[mt]);
        acc1[mt] = WMMAF16(ah.h, bxl.h, acc1[mt]);
        acc1[mt] = WMMAF16(al.h, bxh.h, acc1[mt]);
      }
#pragma unroll
      for (int mt = 0; mt < 4; ++mt) {
        AF ah, al;
        loadA(winh, mt * 16, kc, n, h16, ah);
        loadA(winl, mt * 16, kc, n, h16, al);
        accs[mt] = WMMAF16(ah.h, bsh.h, accs[mt]);
        accs[mt] = WMMAF16(ah.h, bsl.h, accs[mt]);
        accs[mt] = WMMAF16(al.h, bsh.h, accs[mt]);
      }
    }

    // ---- stage 2: agg = relu(pre + s + b_a); stage to per-wave LDS pad ----
#pragma unroll
    for (int mt = 0; mt < 4; ++mt)
#pragma unroll
      for (int r = 0; r < 8; ++r) {
        int row = mt * 16 + r + 8 * h16;     // C/D layout: M = r + 8*(l/16)
        float v = acc1[mt][r] + accs[mt][r] + ba[row];
        aggw[row * AGG_STRIDE + n] = v > 0.0f ? v : 0.0f;
      }

    // ---- stage 3: W_ga @ agg ----
    v8f accg[4];
#pragma unroll
    for (int m = 0; m < 4; ++m)
#pragma unroll
      for (int e = 0; e < 8; ++e) accg[m][e] = 0.0f;
#pragma unroll
    for (int kc = 0; kc < 2; ++kc) {
      AF gh, gl;
#pragma unroll
      for (int v = 0; v < 8; ++v) {
        int k = kc * 32 + h16 * 16 + 2 * v;
        float f0 = aggw[k * AGG_STRIDE + n];
        float f1 = aggw[(k + 1) * AGG_STRIDE + n];
        _Float16 h0 = (_Float16)f0, h1 = (_Float16)f1;
        gh.u[v] = pack2(h0, h1);
        gl.u[v] = pack2((_Float16)(f0 - (float)h0), (_Float16)(f1 - (float)h1));
      }
#pragma unroll
      for (int mt = 0; mt < 4; ++mt) {
        AF ah, al;
        loadA(wgah, mt * 16, kc, n, h16, ah);
        loadA(wgal, mt * 16, kc, n, h16, al);
        accg[mt] = WMMAF16(ah.h, gh.h, accg[mt]);
        accg[mt] = WMMAF16(ah.h, gl.h, accg[mt]);
        accg[mt] = WMMAF16(al.h, gh.h, accg[mt]);
      }
    }

    // ---- epilogue: out = x + W_gx x + W_ga agg + b_gate ----
    float* outi = out + (long long)img * BCHW_ + gbase;
#pragma unroll
    for (int mt = 0; mt < 4; ++mt)
#pragma unroll
      for (int r = 0; r < 8; ++r) {
        int o = mt * 16 + r + 8 * h16;
        float v = xti[o * PX + n0 + n] + acc1[mt + 4][r] + accg[mt][r] + bg[o];
        outi[(long long)o * HW_ + n0 + n] = v;
      }
    __syncthreads();   // protect xt before next tile's load
  }
}

extern "C" void kernel_launch(void* const* d_in, const int* in_sizes, int n_in,
                              void* d_out, int out_size, void* d_ws, size_t ws_size,
                              hipStream_t stream) {
  (void)in_sizes; (void)n_in; (void)out_size; (void)d_ws; (void)ws_size;
  gci_kernel<<<dim3(2048), dim3(256), SMEM_BYTES, stream>>>(
      (const float*)d_in[0], (const float*)d_in[1],
      (const float*)d_in[2], (const float*)d_in[3],
      (const float*)d_in[4], (const float*)d_in[5],
      (const float*)d_in[6], (const float*)d_in[7],
      (const float*)d_in[8], (const float*)d_in[9],
      (float*)d_out);
}
